// MLASelfAttention_84404697301023
// MI455X (gfx1250) — matrix operand order repaired and checked by reading, not compiled, not run on hardware
//
#include <hip/hip_runtime.h>
#include <hip/hip_bf16.h>

typedef __bf16 bf16_t;
typedef __attribute__((ext_vector_type(16))) __bf16 v16bf;
typedef __attribute__((ext_vector_type(8)))  __bf16 v8bf;
typedef __attribute__((ext_vector_type(8)))  float  v8f;

#define S_LEN  2048
#define DMODEL 2048
#define NHEADS 16
#define HDIM   128
#define KVRANK 256

// ---------------------------------------------------------------------------
// WMMA fragment helpers (CDNA5 16x16x32 bf16, wave32)
// A (16xK=32): lane m = lane&15, half = lane>>4 holds K in [8h,8h+8) and [16+8h,24+8h)
// B (K=32x16): lane n = lane&15, half holds K in [8h,8h+8) and [16+8h,24+8h)  (same order as A)
// C (16x16 f32): VGPR r holds row (r + 8*half), col = lane&15
// ---------------------------------------------------------------------------
static __device__ inline v16bf frag_a_load(const bf16_t* base, int ld, int m0, int k0) {
  int lane = threadIdx.x & 31;
  int m  = m0 + (lane & 15);
  int hf = lane >> 4;
  const bf16_t* p = base + (size_t)m * ld + k0 + hf * 8;
  v8bf lo = *reinterpret_cast<const v8bf*>(p);
  v8bf hi = *reinterpret_cast<const v8bf*>(p + 16);
  v16bf f;
#pragma unroll
  for (int i = 0; i < 8; ++i) { f[i] = lo[i]; f[i + 8] = hi[i]; }
  return f;
}

static __device__ inline v16bf frag_b_load(const bf16_t* base, int ld, int n0, int k0) {
  int lane = threadIdx.x & 31;
  int n  = n0 + (lane & 15);
  int hf = lane >> 4;
  const bf16_t* p = base + (size_t)n * ld + k0 + hf * 8;
  v8bf lo = *reinterpret_cast<const v8bf*>(p);
  v8bf hi = *reinterpret_cast<const v8bf*>(p + 16);
  v16bf f;
#pragma unroll
  for (int i = 0; i < 8; ++i) { f[i] = lo[i]; f[i + 8] = hi[i]; }
  return f;
}

static __device__ inline v8f wmma_bf16(v16bf a, v16bf b, v8f c) {
  return __builtin_amdgcn_wmma_f32_16x16x32_bf16(false, a, false, b, (short)0, c,
                                                 false, false);
}

// Async global->LDS copy, 16 bytes per lane (GLOBAL_LOAD_ASYNC_TO_LDS_B128,
// tracked by ASYNCcnt). lds_off = low 32 bits of the generic LDS address
// (aperture rule: generic LDS addr low bits == LDS byte offset).
static __device__ inline void async_copy_b128(unsigned lds_off, const bf16_t* g) {
  asm volatile("global_load_async_to_lds_b128 %0, %1, off"
               :: "v"(lds_off), "v"(g) : "memory");
}
static __device__ inline void wait_asynccnt0() {
  asm volatile("s_wait_asynccnt 0" ::: "memory");
}

// ---------------------------------------------------------------------------
// fp32 -> bf16 cast
// ---------------------------------------------------------------------------
__global__ void cvt_f32_to_bf16(const float* __restrict__ in, bf16_t* __restrict__ out,
                                int n) {
  int i = blockIdx.x * blockDim.x + threadIdx.x;
  if (i < n) out[i] = (bf16_t)in[i];
}

// ---------------------------------------------------------------------------
// C[M,N] (f32) = A[M,K] (bf16, row-major) @ W[N,K]^T (bf16, row-major weights)
// Block: 128 threads = 4 waves; block tile 64x64; K-step 32.
// A/B tiles staged in LDS via async copy engine; fragments read with ds_load.
// ---------------------------------------------------------------------------
__global__ __launch_bounds__(128)
void gemm_bf16_nt(const bf16_t* __restrict__ A, const bf16_t* __restrict__ W,
                  float* __restrict__ C, int M, int N, int K) {
  __shared__ bf16_t As[64 * 32];
  __shared__ bf16_t Bs[64 * 32];

  int tid   = threadIdx.x;
  int wave  = tid >> 5;
  int lane  = tid & 31;
  int mTile = blockIdx.y * 64;
  int nTile = blockIdx.x * 64;
  (void)M;

  // staging assignment: thread -> (row = tid/2, 16-elem chunk = tid&1)
  int crow = tid >> 1;
  int ccol = (tid & 1) * 16;
  const bf16_t* gA = A + (size_t)(mTile + crow) * K + ccol;
  const bf16_t* gW = W + (size_t)(nTile + crow) * K + ccol;
  unsigned ldsA = (unsigned)(size_t)&As[crow * 32 + ccol];
  unsigned ldsB = (unsigned)(size_t)&Bs[crow * 32 + ccol];

  v8f acc[4];
#pragma unroll
  for (int j = 0; j < 4; ++j)
#pragma unroll
    for (int r = 0; r < 8; ++r) acc[j][r] = 0.0f;

  for (int k0 = 0; k0 < K; k0 += 32) {
    // fill LDS tiles with the async copy engine (ASYNCcnt)
    async_copy_b128(ldsA,      gA + k0);
    async_copy_b128(ldsA + 16, gA + k0 + 8);
    async_copy_b128(ldsB,      gW + k0);
    async_copy_b128(ldsB + 16, gW + k0 + 8);
    if (k0 + 32 < K) {  // nudge next tile toward cache
      __builtin_prefetch(gA + k0 + 32, 0, 1);
      __builtin_prefetch(gW + k0 + 32, 0, 1);
    }
    wait_asynccnt0();
    __syncthreads();

    v16bf a = frag_a_load(As, 32, wave * 16, 0);
#pragma unroll
    for (int j = 0; j < 4; ++j) {
      v16bf b = frag_b_load(Bs, 32, j * 16, 0);
      acc[j] = wmma_bf16(a, b, acc[j]);
    }
    __syncthreads();  // WAR: LDS consumed before next fill
  }

  int hf = lane >> 4;
  int n  = lane & 15;
#pragma unroll
  for (int j = 0; j < 4; ++j)
#pragma unroll
    for (int r = 0; r < 8; ++r)
      C[(size_t)(mTile + wave * 16 + r + 8 * hf) * N + nTile + j * 16 + n] = acc[j][r];
}

// ---------------------------------------------------------------------------
// RoPE + head split. q_f32[S,2048], kv_f32[S,4096] -> Qh/Kh [H][S][128] bf16,
// Vt [H][128][S] bf16 (transposed so P*V B-fragments load contiguously).
// ---------------------------------------------------------------------------
__global__ __launch_bounds__(256)
void rope_prep(const float* __restrict__ qf, const float* __restrict__ kvf,
               bf16_t* __restrict__ Qh, bf16_t* __restrict__ Kh,
               bf16_t* __restrict__ Vt) {
  int s = blockIdx.x;
  for (int c = threadIdx.x; c < DMODEL; c += 256) {
    int h = c >> 7, d = c & 127;
    int i = d & 63;
    // inv_freq = ROPE_BASE^(-i/64) = exp(-i * ln(1e5)/64)
    float inv = __expf(-(float)i * (11.512925464970229f / 64.0f));
    float ang = (float)s * inv;
    float cs = __cosf(ang), sn = __sinf(ang);

    float qv = qf[(size_t)s * DMODEL + c];
    float qo = (d < 64) ? -qf[(size_t)s * DMODEL + c + 64]
                        :  qf[(size_t)s * DMODEL + c - 64];
    float kv = kvf[(size_t)s * 2 * DMODEL + c];
    float ko = (d < 64) ? -kvf[(size_t)s * 2 * DMODEL + c + 64]
                        :  kvf[(size_t)s * 2 * DMODEL + c - 64];

    Qh[((size_t)h * S_LEN + s) * HDIM + d] = (bf16_t)(qv * cs + qo * sn);
    Kh[((size_t)h * S_LEN + s) * HDIM + d] = (bf16_t)(kv * cs + ko * sn);

    float vv = kvf[(size_t)s * 2 * DMODEL + DMODEL + c];
    Vt[((size_t)h * HDIM + d) * S_LEN + s] = (bf16_t)vv;
  }
}

// ---------------------------------------------------------------------------
// Causal flash attention, 1 wave = 16 query rows, 4 waves/block.
// scores = Q K^T * 1/sqrt(128) via WMMA; online softmax; O += P V via WMMA.
// Waves are fully independent (causal trip counts differ) -> no block barriers.
// ---------------------------------------------------------------------------
__global__ __launch_bounds__(128)
void mla_attn(const bf16_t* __restrict__ Qh, const bf16_t* __restrict__ Kh,
              const bf16_t* __restrict__ Vt, bf16_t* __restrict__ attn) {
  __shared__ bf16_t ldsP[4 * 16 * 32];  // per-wave 16x32 P staging

  int wave  = threadIdx.x >> 5;
  int lane  = threadIdx.x & 31;
  int h     = blockIdx.y;
  int qBase = (blockIdx.x * 4 + wave) * 16;

  const bf16_t* Q  = Qh + ((size_t)h * S_LEN + qBase) * HDIM;
  const bf16_t* Kb = Kh + (size_t)h * S_LEN * HDIM;
  const bf16_t* Vb = Vt + (size_t)h * HDIM * S_LEN;
  bf16_t* Pw = &ldsP[wave * 16 * 32];

  // Q fragments for the whole row of key tiles (head_dim 128 = 4 k-steps of 32)
  v16bf aq[4];
#pragma unroll
  for (int kk = 0; kk < 4; ++kk) aq[kk] = frag_a_load(Q, HDIM, 0, kk * 32);

  v8f o[8];
#pragma unroll
  for (int j = 0; j < 8; ++j)
#pragma unroll
    for (int r = 0; r < 8; ++r) o[j][r] = 0.0f;

  float rowmax[8], rowsum[8];
#pragma unroll
  for (int r = 0; r < 8; ++r) { rowmax[r] = -1e30f; rowsum[r] = 0.0f; }

  const float scale = 0.08838834764831845f;  // 1/sqrt(128)
  int hf = lane >> 4;
  int n  = lane & 15;

  for (int kb = 0; kb < qBase + 16; kb += 32) {
    // prefetch next key tile
    if (kb + 32 < qBase + 16)
      __builtin_prefetch(Kb + (size_t)(kb + 32) * HDIM, 0, 1);

    // --- scores: two 16-col fragments over 32 keys ---
    v8f s0, s1;
#pragma unroll
    for (int r = 0; r < 8; ++r) { s0[r] = 0.0f; s1[r] = 0.0f; }
#pragma unroll
    for (int kk = 0; kk < 4; ++kk) {
      v16bf b0 = frag_b_load(Kb, HDIM, kb,      kk * 32);
      v16bf b1 = frag_b_load(Kb, HDIM, kb + 16, kk * 32);
      s0 = wmma_bf16(aq[kk], b0, s0);
      s1 = wmma_bf16(aq[kk], b1, s1);
    }

    // --- online softmax (row stats uniform across the 16-lane half) ---
#pragma unroll
    for (int r = 0; r < 8; ++r) {
      int mg = qBase + r + 8 * hf;
      float v0 = s0[r] * scale; if (kb + n      > mg) v0 = -1e30f;
      float v1 = s1[r] * scale; if (kb + 16 + n > mg) v1 = -1e30f;

      float mx = fmaxf(v0, v1);
      mx = fmaxf(mx, __shfl_xor(mx, 1));
      mx = fmaxf(mx, __shfl_xor(mx, 2));
      mx = fmaxf(mx, __shfl_xor(mx, 4));
      mx = fmaxf(mx, __shfl_xor(mx, 8));

      float mNew = fmaxf(rowmax[r], mx);
      float corr = __expf(rowmax[r] - mNew);
      float p0 = __expf(v0 - mNew);
      float p1 = __expf(v1 - mNew);

      float rs = p0 + p1;
      rs += __shfl_xor(rs, 1);
      rs += __shfl_xor(rs, 2);
      rs += __shfl_xor(rs, 4);
      rs += __shfl_xor(rs, 8);

      rowsum[r] = rowsum[r] * corr + rs;
      rowmax[r] = mNew;
#pragma unroll
      for (int j = 0; j < 8; ++j) o[j][r] *= corr;

      int row = r + 8 * hf;
      Pw[row * 32 + n]      = (bf16_t)p0;
      Pw[row * 32 + 16 + n] = (bf16_t)p1;
    }

    asm volatile("s_wait_dscnt 0" ::: "memory");

    // --- O += P @ V (k-dim = 32 keys, 8 head-dim column tiles) ---
    v16bf ap = frag_a_load(Pw, 32, 0, 0);
#pragma unroll
    for (int j = 0; j < 8; ++j) {
      v16bf bv = frag_b_load(Vb, S_LEN, j * 16, kb);
      o[j] = wmma_bf16(ap, bv, o[j]);
    }
  }

  // normalize + write attn (bf16, [S, D_MODEL])
#pragma unroll
  for (int r = 0; r < 8; ++r) {
    float inv = 1.0f / rowsum[r];
#pragma unroll
    for (int j = 0; j < 8; ++j)
      attn[(size_t)(qBase + r + 8 * hf) * DMODEL + h * HDIM + j * 16 + n] =
          (bf16_t)(o[j][r] * inv);
  }
}

// ---------------------------------------------------------------------------
extern "C" void kernel_launch(void* const* d_in, const int* in_sizes, int n_in,
                              void* d_out, int out_size, void* d_ws, size_t ws_size,
                              hipStream_t stream) {
  (void)in_sizes; (void)n_in; (void)out_size; (void)ws_size;
  const float* x         = (const float*)d_in[0];
  const float* w_q       = (const float*)d_in[1];
  const float* w_kv_down = (const float*)d_in[2];
  const float* w_kv_up   = (const float*)d_in[3];
  const float* w_out     = (const float*)d_in[4];
  float* out = (float*)d_out;

  char* ws = (char*)d_ws;
  size_t off = 0;
  auto carve = [&](size_t bytes) -> void* {
    void* p = ws + off;
    off += (bytes + 255) & ~(size_t)255;
    return p;
  };
  bf16_t* x_bf   = (bf16_t*)carve((size_t)S_LEN * DMODEL * 2);
  bf16_t* wq_bf  = (bf16_t*)carve((size_t)DMODEL * DMODEL * 2);
  bf16_t* wdn_bf = (bf16_t*)carve((size_t)KVRANK * DMODEL * 2);
  bf16_t* wup_bf = (bf16_t*)carve((size_t)2 * DMODEL * KVRANK * 2);
  bf16_t* wo_bf  = (bf16_t*)carve((size_t)DMODEL * DMODEL * 2);
  float*  q_f    = (float*)carve((size_t)S_LEN * DMODEL * 4);
  float*  lat_f  = (float*)carve((size_t)S_LEN * KVRANK * 4);
  bf16_t* lat_bf = (bf16_t*)carve((size_t)S_LEN * KVRANK * 2);
  float*  kv_f   = (float*)carve((size_t)S_LEN * 2 * DMODEL * 4);
  bf16_t* Qh     = (bf16_t*)carve((size_t)NHEADS * S_LEN * HDIM * 2);
  bf16_t* Kh     = (bf16_t*)carve((size_t)NHEADS * S_LEN * HDIM * 2);
  bf16_t* Vt     = (bf16_t*)carve((size_t)NHEADS * HDIM * S_LEN * 2);
  bf16_t* attn   = (bf16_t*)carve((size_t)S_LEN * DMODEL * 2);

  auto cvt = [&](const float* src, bf16_t* dst, int nElem) {
    cvt_f32_to_bf16<<<(nElem + 255) / 256, 256, 0, stream>>>(src, dst, nElem);
  };
  cvt(x,         x_bf,   S_LEN * DMODEL);
  cvt(w_q,       wq_bf,  DMODEL * DMODEL);
  cvt(w_kv_down, wdn_bf, KVRANK * DMODEL);
  cvt(w_kv_up,   wup_bf, 2 * DMODEL * KVRANK);
  cvt(w_out,     wo_bf,  DMODEL * DMODEL);

  dim3 blk(128);
  // q = x @ w_q^T
  gemm_bf16_nt<<<dim3(DMODEL / 64, S_LEN / 64), blk, 0, stream>>>(
      x_bf, wq_bf, q_f, S_LEN, DMODEL, DMODEL);
  // latent = x @ w_kv_down^T
  gemm_bf16_nt<<<dim3(KVRANK / 64, S_LEN / 64), blk, 0, stream>>>(
      x_bf, wdn_bf, lat_f, S_LEN, KVRANK, DMODEL);
  cvt(lat_f, lat_bf, S_LEN * KVRANK);
  // kv = latent @ w_kv_up^T
  gemm_bf16_nt<<<dim3(2 * DMODEL / 64, S_LEN / 64), blk, 0, stream>>>(
      lat_bf, wup_bf, kv_f, S_LEN, 2 * DMODEL, KVRANK);
  // RoPE + head split (+ V transpose)
  rope_prep<<<S_LEN, 256, 0, stream>>>(q_f, kv_f, Qh, Kh, Vt);
  // causal flash attention
  mla_attn<<<dim3(S_LEN / 64, NHEADS), 128, 0, stream>>>(Qh, Kh, Vt, attn);
  // out = attn @ w_out^T
  gemm_bf16_nt<<<dim3(DMODEL / 64, S_LEN / 64), blk, 0, stream>>>(
      attn, wo_bf, out, S_LEN, DMODEL, DMODEL);
}
